// S4D_8100308320595
// MI455X (gfx1250) — matrix operand order repaired and checked by reading, not compiled, not run on hardware
//
#include <hip/hip_runtime.h>
#include <hip/hip_bf16.h>

#define BATCH  8
#define DMODEL 256
#define DSTATE 64
#define SEQL   2048
#define KPAD   64
#define KLEN   (SEQL + KPAD)   // 2112: reversed kernel + zero pad for triangular edge
#define SA     2080            // LDS stride per shifted kr copy (mult. of 8, covers max read 2079)
#define NCOPY  8               // shifted copies -> every A-fragment run is 16B aligned

typedef __attribute__((ext_vector_type(16))) __bf16        v16bf;
typedef __attribute__((ext_vector_type(8)))  float         v8f;
typedef __attribute__((ext_vector_type(8)))  unsigned int  uv8;

static __device__ __forceinline__ unsigned short f2bf(float f) {
  unsigned int u = __builtin_bit_cast(unsigned int, f);
  u += 0x7FFFu + ((u >> 16) & 1u);           // round-to-nearest-even
  return (unsigned short)(u >> 16);
}

// Two aligned 16B LDS reads -> one 16x32 bf16 WMMA operand (ds_load_b128 x2).
static __device__ __forceinline__ v16bf ld_frag(const unsigned short* p, int off2) {
  uint4 u0 = *reinterpret_cast<const uint4*>(p);
  uint4 u1 = *reinterpret_cast<const uint4*>(p + off2);
  uv8 u = {u0.x, u0.y, u0.z, u0.w, u1.x, u1.y, u1.z, u1.w};
  return __builtin_bit_cast(v16bf, u);
}

// ---------------------------------------------------------------------------
// Kernel 1: per-(h,n) pole parameters.
//   A = -exp(A_real) - i*A_imag ; dt = exp(inv_dt)
//   dtA = dt*A ; Ceff = (Bc*Cc)*(exp(dtA)-1)/A
// ---------------------------------------------------------------------------
__global__ void s4d_params(const float* __restrict__ A_real,
                           const float* __restrict__ A_imag,
                           const float* __restrict__ B,
                           const float* __restrict__ C,
                           const float* __restrict__ inv_dt,
                           float* __restrict__ dtAr, float* __restrict__ dtAi,
                           float* __restrict__ Cer,  float* __restrict__ Cei) {
  int idx = blockIdx.x * blockDim.x + threadIdx.x;
  if (idx >= DMODEL * DSTATE) return;
  int h = idx / DSTATE;
  float Ar = -__expf(A_real[idx]);
  float Ai = -A_imag[idx];
  float dt = __expf(inv_dt[h]);
  float ar = dt * Ar, ai = dt * Ai;
  float e = __expf(ar);
  float s, c; __sincosf(ai, &s, &c);
  float wr = __fmaf_rn(e, c, -1.0f), wi = e * s;     // exp(dtA)-1
  float Br = B[idx * 2], Bi = B[idx * 2 + 1];
  float Cr = C[idx * 2], Ci = C[idx * 2 + 1];
  float bcr = Br * Cr - Bi * Ci, bci = Br * Ci + Bi * Cr;
  float tr = bcr * wr - bci * wi, ti = bcr * wi + bci * wr;
  float inv = 1.0f / (Ar * Ar + Ai * Ai);            // divide by A via conj(A)/|A|^2
  dtAr[idx] = ar;
  dtAi[idx] = ai;
  Cer[idx]  = (tr * Ar + ti * Ai) * inv;
  Cei[idx]  = (ti * Ar - tr * Ai) * inv;
}

// ---------------------------------------------------------------------------
// Kernel 2: k[h,l] = 2*Re sum_n Ceff * exp(dtA*l), stored REVERSED in bf16:
//   kr[h,t] = k[h, SEQL-1-t] for t<SEQL, 0 for the KPAD tail.
// ---------------------------------------------------------------------------
__global__ __launch_bounds__(256)
void s4d_kernelgen(const float* __restrict__ dtAr, const float* __restrict__ dtAi,
                   const float* __restrict__ Cer,  const float* __restrict__ Cei,
                   unsigned short* __restrict__ kr) {
  __shared__ float par[DSTATE], pai[DSTATE], pcr[DSTATE], pci[DSTATE];
  int h = blockIdx.x;
  for (int n = threadIdx.x; n < DSTATE; n += blockDim.x) {
    par[n] = dtAr[h * DSTATE + n];
    pai[n] = dtAi[h * DSTATE + n];
    pcr[n] = Cer[h * DSTATE + n];
    pci[n] = Cei[h * DSTATE + n];
  }
  __syncthreads();
  for (int t = threadIdx.x; t < KLEN; t += blockDim.x) {
    float val = 0.0f;
    if (t < SEQL) {
      float l = (float)(SEQL - 1 - t);
      float acc = 0.0f;
#pragma unroll 4
      for (int n = 0; n < DSTATE; ++n) {
        float e = __expf(par[n] * l);
        float s, c; __sincosf(pai[n] * l, &s, &c);
        acc = __fmaf_rn(e, __fmaf_rn(pcr[n], c, -pci[n] * s), acc);
      }
      val = 2.0f * acc;
    }
    kr[h * KLEN + t] = f2bf(val);
  }
}

// ---------------------------------------------------------------------------
// Kernel 3: causal conv as block-Toeplitz GEMM, WMMA fed from LDS.
// One workgroup per channel h (8 waves). LDS:
//   lA: 8 shifted copies of kr[h]; copy c holds kr[t] at position t+c, so a
//       lane (row r) reading run start t0a = 2047-16m-r+16*hi with c=(r+1)&7
//       gets a 16B-aligned address -> A fragment = 2x ds_load_b128.
//   lX: bf16 x[h] slice, 16 batch rows (8 real + 8 zero), 2048 each ->
//       B fragment (lanes 0-15 K=0-15, lanes 16-31 K=16-31) = 2x ds_load_b128.
// Each wave carries FOUR accumulators (tiles i, i+8, i+16, i+24) sharing one
// B(j) stream -> 2.5 ds_load_b128 per wmma, 4 independent WMMAs in flight.
// Triangular edge handled by wave-uniform phase loops (EXEC stays all-ones).
// ---------------------------------------------------------------------------
#define STEP(ap, acc)                                                      \
  {                                                                        \
    v16bf a = ld_frag(ap, 8); /* slots 0-7 at +0, 8-15 at +8 */             \
    acc = __builtin_amdgcn_wmma_f32_16x16x32_bf16(                         \
        false, a, false, b, (short)0, acc, false, false);                  \
    ap += 32;                                                              \
  }

__global__ __launch_bounds__(256)
void s4d_conv_wmma(const unsigned short* __restrict__ kr,
                   const float* __restrict__ x,
                   float* __restrict__ out) {
  extern __shared__ unsigned short sm[];
  unsigned short* lA = sm;                      // NCOPY * SA = 16640 bf16
  unsigned short* lX = sm + NCOPY * SA;         // 16 * 2048  = 32768 bf16

  const int h = blockIdx.x;
  const unsigned short* krh = kr + h * KLEN;

  // Build shifted kernel copies (no divisions: one small loop per copy).
  for (int c = 0; c < NCOPY; ++c) {
    for (int q = threadIdx.x; q < SA; q += 256) {
      int t = q - c;
      lA[c * SA + q] = (t >= 0 && t < KLEN) ? krh[t] : (unsigned short)0;
    }
  }
  // Build bf16 x slice, batch padded to 16 with zero rows.
  for (int p = threadIdx.x; p < 16 * SEQL; p += 256) {
    int n = p >> 11, l = p & (SEQL - 1);
    unsigned short v = 0;
    if (n < BATCH) v = f2bf(x[(size_t)n * (DMODEL * SEQL) + (size_t)h * SEQL + l]);
    lX[p] = v;
  }
  __syncthreads();

  const int lane = threadIdx.x & 31;
  const int wv   = threadIdx.x >> 5;
  const int r    = lane & 15;                   // A row / B,D column
  const int hig  = lane >> 4;
  const int c    = (r + 1) & 7;                 // alignment copy for this lane
  // Lane-constant part of the A offset: t0a(i, j=0) = (SEQL-1) - 16i - r + 16*hig
  const int aOff = c * SA + (SEQL - 1) - r + (hig << 4) + c;
  const int bOff = (r << 11) + (hig << 4);

  for (int g = 0; g < 4; ++g) {
    const int ibase = wv + (g << 5);            // tiles ibase, +8, +16, +24
    const unsigned short* ap0 = lA + aOff - (ibase << 4);
    const unsigned short* ap1 = ap0 - 128;
    const unsigned short* ap2 = ap0 - 256;
    const unsigned short* ap3 = ap0 - 384;
    const unsigned short* bp  = lX + bOff;
    v8f acc0 = {}, acc1 = {}, acc2 = {}, acc3 = {};
    const int jm0 = ibase >> 1;                 // jm_u = jm0 + 4u (exact)
    int j = 0;
    for (; j <= jm0; ++j) {                     // all 4 tiles active
      v16bf b = ld_frag(bp, 8); bp += 32;
      STEP(ap0, acc0) STEP(ap1, acc1) STEP(ap2, acc2) STEP(ap3, acc3)
    }
#pragma unroll
    for (int q = 0; q < 4; ++q, ++j) {          // tiles 1..3 active
      v16bf b = ld_frag(bp, 8); bp += 32;
      STEP(ap1, acc1) STEP(ap2, acc2) STEP(ap3, acc3)
    }
#pragma unroll
    for (int q = 0; q < 4; ++q, ++j) {          // tiles 2..3 active
      v16bf b = ld_frag(bp, 8); bp += 32;
      STEP(ap2, acc2) STEP(ap3, acc3)
    }
#pragma unroll
    for (int q = 0; q < 4; ++q, ++j) {          // tile 3 active
      v16bf b = ld_frag(bp, 8); bp += 32;
      STEP(ap3, acc3)
    }
    if (r < BATCH) {                            // fused ReLU store, C/D layout
      float* obase = out + ((size_t)r * DMODEL + h) * SEQL;
#pragma unroll
      for (int v = 0; v < 8; ++v) {
        const int row = (hig << 3) + v;
        obase[((ibase)      << 4) + row] = fmaxf(acc0[v], 0.0f);
        obase[((ibase + 8)  << 4) + row] = fmaxf(acc1[v], 0.0f);
        obase[((ibase + 16) << 4) + row] = fmaxf(acc2[v], 0.0f);
        obase[((ibase + 24) << 4) + row] = fmaxf(acc3[v], 0.0f);
      }
    }
  }
}

// ---------------------------------------------------------------------------
extern "C" void kernel_launch(void* const* d_in, const int* in_sizes, int n_in,
                              void* d_out, int out_size, void* d_ws, size_t ws_size,
                              hipStream_t stream) {
  const float* x      = (const float*)d_in[0];  // (8, 256, 2048)
  const float* A_real = (const float*)d_in[1];  // (256, 64)
  const float* A_imag = (const float*)d_in[2];  // (256, 64)
  const float* B      = (const float*)d_in[3];  // (1, 256, 64, 2)
  const float* C      = (const float*)d_in[4];  // (1, 256, 64, 2)
  const float* inv_dt = (const float*)d_in[5];  // (256, 1)
  float* out = (float*)d_out;                   // (8, 1, 256, 2048)

  const int HN = DMODEL * DSTATE;               // 16384
  float* ws   = (float*)d_ws;                   // ~1.34 MB total scratch
  float* dtAr = ws;
  float* dtAi = ws + HN;
  float* Cer  = ws + 2 * HN;
  float* Cei  = ws + 3 * HN;
  unsigned short* kr = (unsigned short*)(ws + 4 * HN);  // DMODEL * KLEN bf16

  s4d_params<<<(HN + 255) / 256, 256, 0, stream>>>(
      A_real, A_imag, B, C, inv_dt, dtAr, dtAi, Cer, Cei);

  s4d_kernelgen<<<DMODEL, 256, 0, stream>>>(dtAr, dtAi, Cer, Cei, kr);

  // One block per channel; dynamic LDS: 16640 + 32768 bf16 = 98816 bytes
  // (within CDNA5's 320 KB/WGP budget).
  const size_t lds_bytes = (size_t)(NCOPY * SA + 16 * SEQL) * sizeof(unsigned short);
  s4d_conv_wmma<<<DMODEL, 256, lds_bytes, stream>>>(kr, x, out);
}
